// AttentionDCA_45363444580804
// MI455X (gfx1250) — compile-verified, matrix operand order checked by reading, not run on hardware
//
#include <hip/hip_runtime.h>
#include <hip/hip_bf16.h>

// ---------------------------------------------------------------------------
// AttentionDCA pseudo-likelihood on MI455X (gfx1250, wave32, WMMA + TDM).
//
// Shapes: H=32 heads, L=256 positions, DK=32, q=21 (pad 32), D_IN=64, M=4096.
// Stage 4 (mat_ene) dominates: one-hot einsum recast as bf16 WMMA GEMM with
// the one-hot operand synthesized from Z in registers; J (128MB bf16) lives
// in L2 and is staged into LDS by the Tensor Data Mover.
// ---------------------------------------------------------------------------

typedef __attribute__((ext_vector_type(16))) __bf16 v16bf;
typedef __attribute__((ext_vector_type(8)))  __bf16 v8bf;
typedef __attribute__((ext_vector_type(8)))  float  v8f;
typedef __attribute__((ext_vector_type(4)))  unsigned int u32x4;
typedef __attribute__((ext_vector_type(8)))  int    i32x8;
typedef __attribute__((ext_vector_type(4)))  int    i32x4;

#define H_    32
#define L_    256
#define DK_   32
#define QA_   21
#define QPAD  32
#define M_    4096
#define LAMBDA_ 1e-3f

#if defined(__has_builtin)
#if __has_builtin(__builtin_amdgcn_tensor_load_to_lds) && __has_builtin(__builtin_amdgcn_s_wait_tensorcnt)
#define HAVE_TDM 1
#endif
#endif
#ifndef HAVE_TDM
#define HAVE_TDM 0
#endif

// Workspace layout (bytes)
static constexpr size_t OFF_P    = 0;                                   // P probs: 32*256*256*4 = 8 MB
static constexpr size_t OFF_V1   = 8ull * 1024 * 1024;                  // V1: 32*21*32*4 = 84 KB
static constexpr size_t OFF_VAA  = OFF_V1 + 96 * 1024;                  // Vaa bf16 padded: 32*1024*2 = 64 KB
static constexpr size_t OFF_JB   = OFF_VAA + 64 * 1024;                 // J bf16 padded: 65536*1024*2 = 128 MB
static constexpr size_t OFF_REGP = OFF_JB + (size_t)65536 * 1024 * 2;   // 4096 floats
static constexpr size_t OFF_PLP  = OFF_REGP + 16384;                    // 4096 floats

// 16-bit A/B fragment K-index mapping for V_WMMA_*_16X16X32 (ISA 7.12.2):
// lanes 0-15 hold K in {0..7, 16..23}, lanes 16-31 hold K in {8..15, 24..31}.
__device__ __forceinline__ int kmap16(int lane, int e) {
    int khalf = lane >> 4;
    return (e < 8) ? (khalf * 8 + e) : (16 + khalf * 8 + (e - 8));
}

#if HAVE_TDM
// Build a D# (ISA 08_async_tensor §8) for a 2D tile and fire the TDM.
// group0: count=1 | lds_addr | global_addr(57b) | type=2 ("image").
// group1: data_size | tensor_dim0/1 | tile_dim0/1 | tensor_dim0_stride.
// Groups 2/3 zero (tensor <= 2D). This toolchain's builtin takes 6 args
// (clang-23 / therock-10.0 form): (g0, g1, g2, g3, g_extra, cpol).
__device__ __forceinline__ void tdm_load_to_lds(unsigned lds_off, const void* gsrc,
                                                unsigned dsize_code,
                                                unsigned tensor_d0, unsigned tensor_d1,
                                                unsigned tile_d0, unsigned tile_d1,
                                                unsigned long long stride0) {
    unsigned long long ga = (unsigned long long)(uintptr_t)gsrc;
    u32x4 g0;
    g0[0] = 1u;                                                  // count=1, user mode
    g0[1] = lds_off;                                             // lds_addr [63:32]
    g0[2] = (unsigned)ga;                                        // global_addr lo
    g0[3] = (unsigned)((ga >> 32) & 0x01FFFFFFu) | 0x80000000u;  // addr hi | type=2
    i32x8 g1;
    g1[0] = (int)(dsize_code << 16);                             // data_size @ [17:16]
    g1[1] = (int)((tensor_d0 & 0xFFFFu) << 16);                  // tensor_dim0 [79:48]
    g1[2] = (int)(((tensor_d0 >> 16) & 0xFFFFu) |
                  ((tensor_d1 & 0xFFFFu) << 16));                // | tensor_dim1 [111:80]
    g1[3] = (int)(((tensor_d1 >> 16) & 0xFFFFu) |
                  ((tile_d0 & 0xFFFFu) << 16));                  // | tile_dim0 [127:112]
    g1[4] = (int)(tile_d1 & 0xFFFFu);                            // tile_dim1 [143:128]
    g1[5] = (int)(stride0 & 0xFFFFFFFFu);                        // dim0_stride [207:160]
    g1[6] = (int)((stride0 >> 32) & 0xFFFFu);
    g1[7] = 0;
    i32x4 gz4 = {0, 0, 0, 0};
    i32x8 gz8 = {0, 0, 0, 0, 0, 0, 0, 0};
    __builtin_amdgcn_tensor_load_to_lds(g0, g1, gz4, gz4, gz8, 0);
}
#endif

// -------------------- Kernel 1: scores = QK^T/sqrt(dk), row softmax --------
__global__ __launch_bounds__(256) void k1_attn(const float* __restrict__ Q,
                                               const float* __restrict__ K,
                                               float* __restrict__ P) {
    const int h   = blockIdx.y;
    const int i0  = blockIdx.x * 32;
    const int tid = threadIdx.x;
    const int lane = tid & 31, w = tid >> 5;
    __shared__ float S[32 * 256];                       // 32 KB
    const float scale = 0.17677669529663687f;           // 1/sqrt(32)
    const int mrow = lane & 15;

    v16bf afrag[2];
    for (int mt = 0; mt < 2; ++mt) {
        const float* qrow = Q + ((size_t)h * L_ + (i0 + mt * 16 + mrow)) * DK_;
        for (int e = 0; e < 16; ++e)
            afrag[mt][e] = (__bf16)(qrow[kmap16(lane, e)] * scale);
    }
    for (int t = 0; t < 2; ++t) {
        const int nt = w * 2 + t;
        const float* krow = K + ((size_t)h * L_ + (nt * 16 + (lane & 15))) * DK_;
        v16bf bfrag;
        for (int e = 0; e < 16; ++e) bfrag[e] = (__bf16)krow[kmap16(lane, e)];
        for (int mt = 0; mt < 2; ++mt) {
            v8f c = {0.f, 0.f, 0.f, 0.f, 0.f, 0.f, 0.f, 0.f};
            c = __builtin_amdgcn_wmma_f32_16x16x32_bf16(false, afrag[mt], false, bfrag,
                                                        (short)0, c, false, false);
            for (int v = 0; v < 8; ++v) {
                int row = mt * 16 + ((lane < 16) ? v : v + 8);
                int col = nt * 16 + (lane & 15);
                S[row * 256 + col] = c[v];
            }
        }
    }
    __syncthreads();
    const int row = tid >> 3, l8 = tid & 7;
    float mx = -3.4e38f;
    for (int c = l8; c < 256; c += 8) mx = fmaxf(mx, S[row * 256 + c]);
    for (int d = 4; d >= 1; d >>= 1) mx = fmaxf(mx, __shfl_xor(mx, d, 8));
    float s = 0.f;
    for (int c = l8; c < 256; c += 8) s += __expf(S[row * 256 + c] - mx);
    for (int d = 4; d >= 1; d >>= 1) s += __shfl_xor(s, d, 8);
    const float inv = 1.0f / s;
    float* prow = P + ((size_t)h * L_ + (i0 + row)) * L_;
    for (int c = l8; c < 256; c += 8) prow[c] = __expf(S[row * 256 + c] - mx) * inv;
}

// -------------------- Kernel 2: V_aa RBF kernel, padded to 32x32 -----------
__global__ __launch_bounds__(256) void k2_vaa(const float* __restrict__ Vm,
                                              const float* __restrict__ reps,
                                              float* __restrict__ V1,
                                              __bf16* __restrict__ VaaBf) {
    const int tid = threadIdx.x;
    for (int idx = tid; idx < H_ * QA_ * 32; idx += 256) {
        int h = idx / (QA_ * 32);
        int rem = idx % (QA_ * 32);
        int qq = rem / 32, e = rem % 32;
        float acc = 0.f;
        const float* rrow = reps + qq * 64;
        const float* vcol = Vm + (size_t)h * 64 * 32 + e;
        for (int d = 0; d < 64; ++d) acc += rrow[d] * vcol[d * 32];
        V1[idx] = acc;
    }
    __syncthreads();
    for (int idx = tid; idx < H_ * QPAD * QPAD; idx += 256) {
        int h = idx >> 10;
        int b = (idx >> 5) & 31, a = idx & 31;
        float val = 0.f;
        if (b < QA_ && a < QA_) {
            const float* vb = V1 + ((size_t)h * QA_ + b) * 32;
            const float* va = V1 + ((size_t)h * QA_ + a) * 32;
            float sb = 0.f, sa = 0.f, dot = 0.f;
            for (int e = 0; e < 32; ++e) {
                float x = vb[e], y = va[e];
                sb += x * x; sa += y * y; dot += x * y;
            }
            float d2 = fmaxf(sb + sa - 2.f * dot, 0.f);
            val = __expf(-d2 * (1.0f / 21.0f));
        }
        VaaBf[idx] = (__bf16)val;
    }
}

// -------------------- Kernel 3: J[(i,j)][(b,a)] = A·Vaa, GEMM K=32 ---------
__global__ __launch_bounds__(256) void k3_J(const float* __restrict__ P,
                                            const __bf16* __restrict__ VaaBf,
                                            __bf16* __restrict__ Jb,
                                            float* __restrict__ regpart) {
    const int tid = threadIdx.x, lane = tid & 31, w = tid >> 5;
    const int rowbase = blockIdx.x * 16;
    __shared__ float red[256];
    const int mrow = lane & 15;
    const int p = rowbase + mrow;
    const int i = p >> 8, j = p & 255;

    v16bf afrag;                                         // A[(i,j),h] symmetrized
    for (int e = 0; e < 16; ++e) {
        int h = kmap16(lane, e);
        float v = 0.5f * (P[(((size_t)h * L_) + i) * L_ + j] +
                          P[(((size_t)h * L_) + j) * L_ + i]);
        afrag[e] = (__bf16)v;
    }
    float regacc = 0.f;
    for (int ct = 0; ct < 8; ++ct) {
        const int colbase = (w * 8 + ct) * 16;
        const int col = colbase + (lane & 15);
        v16bf bfrag;
        for (int e = 0; e < 16; ++e) bfrag[e] = VaaBf[(size_t)kmap16(lane, e) * 1024 + col];
        v8f c = {0.f, 0.f, 0.f, 0.f, 0.f, 0.f, 0.f, 0.f};
        c = __builtin_amdgcn_wmma_f32_16x16x32_bf16(false, afrag, false, bfrag,
                                                    (short)0, c, false, false);
        for (int v = 0; v < 8; ++v) {
            int prow = rowbase + ((lane < 16) ? v : v + 8);
            float val = c[v];
            if ((prow >> 8) == (prow & 255)) val = 0.f;  // position diagonal
            else regacc += val * val;
            Jb[(size_t)prow * 1024 + col] = (__bf16)val;
        }
    }
    red[tid] = regacc; __syncthreads();
    for (int s = 128; s > 0; s >>= 1) { if (tid < s) red[tid] += red[tid + s]; __syncthreads(); }
    if (tid == 0) regpart[blockIdx.x] = red[0];
}

// -------------------- Kernel 4: mat_ene via one-hot WMMA GEMM + fused lse --
// grid (16 m-chunks, 256 r). Block: 32 b-rows x 256 m cols, K=(j,a)=8192.
// TDM stages J (1D, 32KB) and Z (2D strided tile) into LDS each j-chunk.
__global__ __launch_bounds__(256) void k4_ene(const __bf16* __restrict__ Jb,
                                              const int* __restrict__ Z,
                                              const float* __restrict__ weights,
                                              float* __restrict__ plpart) {
    const int r = blockIdx.y;
    const int m0 = blockIdx.x * 256;
    const int tid = threadIdx.x, lane = tid & 31, w = tid >> 5;
    __shared__ __align__(16) char smem[49152];
    __bf16* JbL = (__bf16*)smem;                 // 16 j x 1024 bf16 = 32 KB @ lds 0
    int*    ZL  = (int*)(smem + 32768);          // 16 j x 256 m    = 16 KB @ lds 32768
    float*  ene = (float*)smem;                  // reuse: 32 b x 256 m f32
    float*  red = (float*)(smem + 32768);        // reuse: 256 f32

    const int mrow  = lane & 15;
    const int khalf = lane >> 4;
    const int abase0 = khalf * 8;                // a-range of frag elems 0..7
    const int abase1 = 16 + khalf * 8;           // a-range of frag elems 8..15

    v8f acc[2][2];
    for (int a = 0; a < 2; ++a)
        for (int b = 0; b < 2; ++b)
            acc[a][b] = (v8f){0.f, 0.f, 0.f, 0.f, 0.f, 0.f, 0.f, 0.f};

    for (int jc = 0; jc < 16; ++jc) {
        __syncthreads();
#if HAVE_TDM
        if (w == 0) {   // one TDM issue per workgroup (EXEC ignored by TDM)
            // J[r] rows jc*16..+16 : contiguous 16384 x 2B
            tdm_load_to_lds(0u, Jb + ((size_t)r * L_ + jc * 16) * 1024,
                            /*dsize=*/1u, 16384u, 1u, 16384u, 1u, 16384ull);
            // Z tile: 16 rows x 256 cols of 4B, row stride M
            tdm_load_to_lds(32768u, Z + (size_t)(jc * 16) * M_ + m0,
                            /*dsize=*/2u, (unsigned)M_, (unsigned)L_,
                            256u, 16u, (unsigned long long)M_);
            __builtin_amdgcn_s_wait_tensorcnt(0);
        }
#else
        {
            const uint4* __restrict__ Jg = (const uint4*)Jb;
            const size_t base16 = ((size_t)r * L_ + jc * 16) * 1024 / 8;
            uint4* dst = (uint4*)JbL;
            for (int t = tid; t < 2048; t += 256) dst[t] = Jg[base16 + t];
            for (int jj = 0; jj < 16; ++jj)
                ZL[jj * 256 + tid] = Z[(size_t)(jc * 16 + jj) * M_ + m0 + tid];
            if (jc < 15 && tid == 0)
                __builtin_prefetch(Jb + ((size_t)r * L_ + (jc + 1) * 16) * 1024, 0, 1);
        }
#endif
        __syncthreads();

        for (int jj = 0; jj < 16; ++jj) {
            // A-fragments: two contiguous 16B runs per lane -> ds_load_b128 x2
            v16bf afrag[2];
            for (int mt = 0; mt < 2; ++mt) {
                const __bf16* src = JbL + jj * 1024 + (mt * 16 + mrow) * 32;
                v8bf lo = *(const v8bf*)(src + abase0);
                v8bf hi = *(const v8bf*)(src + abase1);
                afrag[mt] = __builtin_shufflevector(lo, hi,
                    0, 1, 2, 3, 4, 5, 6, 7, 8, 9, 10, 11, 12, 13, 14, 15);
            }
            // B-fragments: one-hot of Z built as packed bf16 dwords (exact)
            v16bf bfrag[2];
            for (int nt = 0; nt < 2; ++nt) {
                const int mloc = w * 32 + nt * 16 + mrow;
                const int z = ZL[jj * 256 + mloc];
                const int t0 = z - abase0;
                const int t1 = z - abase1;
                const int sh0 = 0x3F80 << ((t0 & 1) << 4);
                const int sh1 = 0x3F80 << ((t1 & 1) << 4);
                i32x8 bi;
                bi[0] = ((t0 >> 1) == 0) ? sh0 : 0;
                bi[1] = ((t0 >> 1) == 1) ? sh0 : 0;
                bi[2] = ((t0 >> 1) == 2) ? sh0 : 0;
                bi[3] = ((t0 >> 1) == 3) ? sh0 : 0;
                bi[4] = ((t1 >> 1) == 0) ? sh1 : 0;
                bi[5] = ((t1 >> 1) == 1) ? sh1 : 0;
                bi[6] = ((t1 >> 1) == 2) ? sh1 : 0;
                bi[7] = ((t1 >> 1) == 3) ? sh1 : 0;
                bfrag[nt] = __builtin_bit_cast(v16bf, bi);
            }
            for (int mt = 0; mt < 2; ++mt)
                for (int nt = 0; nt < 2; ++nt)
                    acc[mt][nt] = __builtin_amdgcn_wmma_f32_16x16x32_bf16(
                        false, afrag[mt], false, bfrag[nt], (short)0, acc[mt][nt],
                        false, false);
        }
    }
    __syncthreads();
    for (int mt = 0; mt < 2; ++mt)
        for (int nt = 0; nt < 2; ++nt)
            for (int v = 0; v < 8; ++v) {
                int b = mt * 16 + ((lane < 16) ? v : v + 8);
                int mloc = w * 32 + nt * 16 + (lane & 15);
                ene[b * 256 + mloc] = acc[mt][nt][v];
            }
    __syncthreads();
    // fused logsumexp over b<21, correct gather, weighted sum
    const int z = Z[(size_t)r * M_ + m0 + tid];
    float mx = -3.4e38f;
    for (int b = 0; b < QA_; ++b) mx = fmaxf(mx, ene[b * 256 + tid]);
    float s = 0.f;
    for (int b = 0; b < QA_; ++b) s += __expf(ene[b * 256 + tid] - mx);
    const float lse = mx + __logf(s);
    const float corr = ene[z * 256 + tid];
    const float contrib = weights[m0 + tid] * (corr - lse);
    __syncthreads();
    red[tid] = contrib; __syncthreads();
    for (int st = 128; st > 0; st >>= 1) { if (tid < st) red[tid] += red[tid + st]; __syncthreads(); }
    if (tid == 0) plpart[blockIdx.y * 16 + blockIdx.x] = red[0];
}

// -------------------- Kernel 5: final scalar ------------------------------
__global__ __launch_bounds__(256) void k5_final(const float* __restrict__ regpart,
                                                const float* __restrict__ plpart,
                                                float* __restrict__ out) {
    __shared__ float red[256];
    const int tid = threadIdx.x;
    float racc = 0.f, pacc = 0.f;
    for (int idx = tid; idx < 4096; idx += 256) { racc += regpart[idx]; pacc += plpart[idx]; }
    red[tid] = racc; __syncthreads();
    for (int s = 128; s > 0; s >>= 1) { if (tid < s) red[tid] += red[tid + s]; __syncthreads(); }
    const float regsum = red[0]; __syncthreads();
    red[tid] = pacc; __syncthreads();
    for (int s = 128; s > 0; s >>= 1) { if (tid < s) red[tid] += red[tid + s]; __syncthreads(); }
    if (tid == 0) out[0] = -red[0] + LAMBDA_ * regsum;
}

extern "C" void kernel_launch(void* const* d_in, const int* in_sizes, int n_in,
                              void* d_out, int out_size, void* d_ws, size_t ws_size,
                              hipStream_t stream) {
    (void)in_sizes; (void)n_in; (void)out_size; (void)ws_size;
    const float* Q    = (const float*)d_in[0];
    const float* K    = (const float*)d_in[1];
    const float* Vm   = (const float*)d_in[2];
    const float* reps = (const float*)d_in[3];
    const float* wts  = (const float*)d_in[4];
    const int*   Z    = (const int*)d_in[5];

    char* ws = (char*)d_ws;
    float*  P       = (float*)(ws + OFF_P);
    float*  V1      = (float*)(ws + OFF_V1);
    __bf16* VaaBf   = (__bf16*)(ws + OFF_VAA);
    __bf16* Jb      = (__bf16*)(ws + OFF_JB);
    float*  regpart = (float*)(ws + OFF_REGP);
    float*  plpart  = (float*)(ws + OFF_PLP);

    k1_attn<<<dim3(8, 32), 256, 0, stream>>>(Q, K, P);
    k2_vaa<<<1, 256, 0, stream>>>(Vm, reps, V1, VaaBf);
    k3_J<<<4096, 256, 0, stream>>>(P, VaaBf, Jb, regpart);
    k4_ene<<<dim3(16, 256), 256, 0, stream>>>(Jb, Z, wts, plpart);
    k5_final<<<1, 256, 0, stream>>>(regpart, plpart, (float*)d_out);
}
